// MemoryRamTwoStreamModule_25623774888600
// MI455X (gfx1250) — matrix operand order repaired and checked by reading, not compiled, not run on hardware
//
#include <hip/hip_runtime.h>
#include <hip/hip_bf16.h>
#include <cstddef>
#include <cstdint>

// ---------------------------------------------------------------------------
// MemoryRamTwoStreamModule — CDNA5 (gfx1250) implementation.
// Weights converted once to bf16 (~115MB) so they stay resident in the 192MB
// L2 across all 4096 sequential steps; per-step GEMVs use
// v_wmma_f32_16x16x32_bf16 with weights pre-packed into the per-lane WMMA
// B-fragment layout (32 contiguous bytes / lane / k-tile -> 2x b128 loads).
// Concat inputs are handled as straight-line per-segment sub-loops (all
// segments are 2048 elems = 64 k-tiles), with dual accumulators to decouple
// consecutive WMMAs.
// ---------------------------------------------------------------------------

#define DIM_I 2048
#define DIM_H 2048
#define DIM_M 512
#define DIM_T 4096

typedef __attribute__((ext_vector_type(16))) __bf16 v16bf;
typedef __attribute__((ext_vector_type(8)))  __bf16 v8bf;
typedef __attribute__((ext_vector_type(8)))  float  v8f;

// ---------------------------------------------------------------------------
// Prep kernels (run once per launch)
// ---------------------------------------------------------------------------

// Pack fp32 row-major W[K x N] into bf16 WMMA B-fragment tiles.
// Tile (nt,kt) = 512 bf16, lane-major: element (lane,j) holds
// W[kt*32 + ((lane&16)>>1) + (j&7) + ((j&8)<<1) , nt*16 + (lane&15)].  kt tiles contiguous per nt
// so a consumer wave streams its whole column strip contiguously.
__global__ void pack_weight_kernel(const float* __restrict__ src,
                                   __bf16* __restrict__ dst,
                                   int K, int N) {
  long total = (long)K * N;
  long idx = (long)blockIdx.x * blockDim.x + threadIdx.x;
  if (idx >= total) return;
  int  j    = (int)(idx & 15);
  int  lane = (int)((idx >> 4) & 31);
  long tile = idx >> 9;
  int  ktiles = K >> 5;
  int  nt = (int)(tile / ktiles);
  int  kt = (int)(tile % ktiles);
  int  k  = kt * 32 + ((lane & 16) >> 1) + (j & 7) + ((j & 8) << 1);
  int  n  = nt * 16 + (lane & 15);
  dst[idx] = (__bf16)src[(long)k * N + n];
}

__global__ void f32_to_bf16_kernel(const float* __restrict__ s,
                                   __bf16* __restrict__ d, long n) {
  long i = (long)blockIdx.x * blockDim.x + threadIdx.x;
  long stride = (long)gridDim.x * blockDim.x;
  for (; i < n; i += stride) d[i] = (__bf16)s[i];
}

__global__ void zero_f32_kernel(float* p, long n) {
  long i = (long)blockIdx.x * blockDim.x + threadIdx.x;
  long stride = (long)gridDim.x * blockDim.x;
  for (; i < n; i += stride) p[i] = 0.f;
}

__global__ void zero_bf16_kernel(__bf16* p, long n) {
  long i = (long)blockIdx.x * blockDim.x + threadIdx.x;
  long stride = (long)gridDim.x * blockDim.x;
  for (; i < n; i += stride) p[i] = (__bf16)0.f;
}

// ---------------------------------------------------------------------------
// One contiguous input segment: `tiles` k-tiles (tiles is even), streaming
// both the x-fragment (broadcast into all 16 A rows) and the packed B strip.
// Two accumulators break the WMMA C-operand dependency chain.
// ---------------------------------------------------------------------------
__device__ __forceinline__ void gemv_seg(const __bf16*& wptr,
                                         const __bf16* __restrict__ sp,
                                         int tiles, int aoff,
                                         v8f& c0, v8f& c1) {
  for (int kt = 0; kt < tiles; kt += 2) {
    union { v16bf v; v8bf h[2]; } a0, a1;
    a0.h[0] = *(const v8bf*)(sp + aoff);
    a0.h[1] = *(const v8bf*)(sp + aoff + 16);
    a1.h[0] = *(const v8bf*)(sp + aoff + 32);
    a1.h[1] = *(const v8bf*)(sp + aoff + 48);
    v16bf b0 = *(const v16bf*)wptr;
    v16bf b1 = *(const v16bf*)(wptr + 512);
    c0 = __builtin_amdgcn_wmma_f32_16x16x32_bf16(
        false, a0.v, false, b0, (short)0, c0, false, false);
    c1 = __builtin_amdgcn_wmma_f32_16x16x32_bf16(
        false, a1.v, false, b1, (short)0, c1, false, false);
    sp += 64;
    wptr += 1024;
  }
}

// One wave computes 16 outputs of y = x @ W (+bias, opt. ReLU); x is the
// concatenation of up to 3 segments, each a multiple of 64 elements.
__device__ __forceinline__ void gemv_wave(
    const __bf16* __restrict__ Wp,   // packed weight base
    const float*  __restrict__ bias,
    const __bf16* s0, int l0,
    const __bf16* s1, int l1,
    const __bf16* s2, int l2,
    int K, int ntile, int act,
    float* outF, __bf16* outB) {
  int lane = threadIdx.x & 31;
  int ktiles = K >> 5;
  const __bf16* wptr = Wp + ((long)ntile * ktiles) * 512 + lane * 16;
  int aoff = (lane & 16) >> 1;     // lanes 0-15: k {0..7,16..23}; 16-31: {8..15,24..31}
  v8f c0 = {}, c1 = {};
  gemv_seg(wptr, s0, l0 >> 5, aoff, c0, c1);
  if (s1) gemv_seg(wptr, s1, l1 >> 5, aoff, c0, c1);
  if (s2) gemv_seg(wptr, s2, l2 >> 5, aoff, c0, c1);
  if (lane < 16) {                 // D row 0: lanes 0-15 = y[ntile*16 + lane]
    int n = ntile * 16 + lane;
    float y = c0[0] + c1[0] + bias[n];
    if (act) y = fmaxf(y, 0.f);
    if (outF) outF[n] = y;
    if (outB) outB[n] = (__bf16)y;
  }
}

// ---------------------------------------------------------------------------
// K1: state-only GEMVs.  Waves 0..351 across blocks 0..43 (8 waves/block):
//   [  0, 32): z_rp = cat3 @ W_rp            (K=6144, N=512)
//   [ 32,160): ca   = relu([ha,xa] @ W_ca)   (K=4096, N=2048)
//   [160,288): cm   = relu([hm,xm] @ W_cm)
//   [288,320): z_wa = [ha,xa] @ W_wa         (K=4096, N=512)
//   [320,352): z_wm = [hm,xm] @ W_wm
// Block 44: z_wp = cat3 @ W_wp (N=3, plain VALU).
// ---------------------------------------------------------------------------
__global__ void k1_kernel(
    const __bf16* __restrict__ ha, const __bf16* __restrict__ hm,
    const __bf16* __restrict__ h,
    const __bf16* __restrict__ xa, const __bf16* __restrict__ xm,
    const __bf16* __restrict__ pW_rp, const __bf16* __restrict__ pW_ca,
    const __bf16* __restrict__ pW_cm, const __bf16* __restrict__ pW_wa,
    const __bf16* __restrict__ pW_wm,
    const float* __restrict__ b_rp, const float* __restrict__ b_ca,
    const float* __restrict__ b_cm, const float* __restrict__ b_wa,
    const float* __restrict__ b_wm,
    const float* __restrict__ W_wp, const float* __restrict__ b_wp,
    float* z_rp, float* ca, float* cm, float* z_wa, float* z_wm, float* z_wp) {
  __shared__ float red[768];
  if (blockIdx.x == 44) {
    int t = threadIdx.x;
    float p0 = 0.f, p1 = 0.f, p2 = 0.f;
    for (int k = t; k < 3 * DIM_H; k += 256) {
      float x = (float)(k < DIM_H ? ha[k]
                 : (k < 2 * DIM_H ? hm[k - DIM_H] : h[k - 2 * DIM_H]));
      p0 += x * W_wp[k * 3 + 0];
      p1 += x * W_wp[k * 3 + 1];
      p2 += x * W_wp[k * 3 + 2];
    }
    red[t] = p0; red[256 + t] = p1; red[512 + t] = p2;
    __syncthreads();
    for (int s = 128; s > 0; s >>= 1) {
      if (t < s) {
        red[t]       += red[t + s];
        red[256 + t] += red[256 + t + s];
        red[512 + t] += red[512 + t + s];
      }
      __syncthreads();
    }
    if (t < 3) z_wp[t] = red[t * 256] + b_wp[t];
    return;
  }
  int wid = blockIdx.x * 8 + (threadIdx.x >> 5);
  if (wid < 32)
    gemv_wave(pW_rp, b_rp, ha, DIM_H, hm, DIM_H, h, DIM_H,
              3 * DIM_H, wid, 0, z_rp, nullptr);
  else if (wid < 160)
    gemv_wave(pW_ca, b_ca, ha, DIM_H, xa, DIM_I, nullptr, 0,
              DIM_H + DIM_I, wid - 32, 1, ca, nullptr);
  else if (wid < 288)
    gemv_wave(pW_cm, b_cm, hm, DIM_H, xm, DIM_I, nullptr, 0,
              DIM_H + DIM_I, wid - 160, 1, cm, nullptr);
  else if (wid < 320)
    gemv_wave(pW_wa, b_wa, ha, DIM_H, xa, DIM_I, nullptr, 0,
              DIM_H + DIM_I, wid - 288, 0, z_wa, nullptr);
  else
    gemv_wave(pW_wm, b_wm, hm, DIM_H, xm, DIM_I, nullptr, 0,
              DIM_H + DIM_I, wid - 320, 0, z_wm, nullptr);
}

// ---------------------------------------------------------------------------
// K2: each of 8 blocks redundantly computes softmax(z_rp) in LDS, then
// r[n] = sum_m ar[m]*mem[m][n] (coalesced fp32 reads), emitting bf16 r.
// ---------------------------------------------------------------------------
__global__ void k2_kernel(const float* __restrict__ z_rp,
                          const float* __restrict__ memv,
                          __bf16* __restrict__ r_bf) {
  __shared__ float sm[512];
  __shared__ float red[256];
  int t = threadIdx.x;
  float v0 = z_rp[t], v1 = z_rp[t + 256];
  red[t] = fmaxf(v0, v1);
  __syncthreads();
  for (int s = 128; s > 0; s >>= 1) {
    if (t < s) red[t] = fmaxf(red[t], red[t + s]);
    __syncthreads();
  }
  float mx = red[0];
  __syncthreads();
  float e0 = __expf(v0 - mx), e1 = __expf(v1 - mx);
  red[t] = e0 + e1;
  __syncthreads();
  for (int s = 128; s > 0; s >>= 1) {
    if (t < s) red[t] += red[t + s];
    __syncthreads();
  }
  float inv = 1.f / red[0];
  sm[t] = e0 * inv; sm[t + 256] = e1 * inv;
  __syncthreads();
  int n = blockIdx.x * 256 + t;
  float acc = 0.f;
#pragma unroll 8
  for (int m = 0; m < DIM_M; ++m)
    acc += sm[m] * memv[(long)m * DIM_H + n];
  r_bf[n] = (__bf16)acc;
}

// ---------------------------------------------------------------------------
// K3: blocks 0..47 -> 384 WMMA waves:
//   [  0,128): h1  = relu([r,h]    @ W_r0) -> d_out row + bf16 state
//   [128,256): ha1 = relu([xa,r,ha]@ W_ra) -> bf16 state
//   [256,384): hm1 = relu([xm,r,hm]@ W_rm) -> bf16 state
// Blocks 48..111: in-place mem update; write softmaxes recomputed in LDS.
// ---------------------------------------------------------------------------
__global__ void k3_kernel(
    const __bf16* __restrict__ r,  const __bf16* __restrict__ h,
    const __bf16* __restrict__ ha, const __bf16* __restrict__ hm,
    const __bf16* __restrict__ xa, const __bf16* __restrict__ xm,
    const __bf16* __restrict__ pW_r0, const __bf16* __restrict__ pW_ra,
    const __bf16* __restrict__ pW_rm,
    const float* __restrict__ b_r0, const float* __restrict__ b_ra,
    const float* __restrict__ b_rm,
    const float* __restrict__ z_wp, const float* __restrict__ z_wa,
    const float* __restrict__ z_wm,
    const float* __restrict__ ca, const float* __restrict__ cm,
    float* __restrict__ memv,
    float* __restrict__ out_h,
    __bf16* __restrict__ h_nxt, __bf16* __restrict__ ha_nxt,
    __bf16* __restrict__ hm_nxt) {
  if (blockIdx.x < 48) {
    int wid = blockIdx.x * 8 + (threadIdx.x >> 5);
    if (wid < 128)
      gemv_wave(pW_r0, b_r0, r, DIM_H, h, DIM_H, nullptr, 0,
                2 * DIM_H, wid, 1, out_h, h_nxt);
    else if (wid < 256)
      gemv_wave(pW_ra, b_ra, xa, DIM_I, r, DIM_H, ha, DIM_H,
                DIM_I + 2 * DIM_H, wid - 128, 1, nullptr, ha_nxt);
    else
      gemv_wave(pW_rm, b_rm, xm, DIM_I, r, DIM_H, hm, DIM_H,
                DIM_I + 2 * DIM_H, wid - 256, 1, nullptr, hm_nxt);
    return;
  }
  // ---- memory update path ----
  __shared__ float sa[512], sb[512], red[256];
  int t = threadIdx.x;
  // softmax(z_wa) -> sa
  {
    float v0 = z_wa[t], v1 = z_wa[t + 256];
    red[t] = fmaxf(v0, v1); __syncthreads();
    for (int s = 128; s > 0; s >>= 1) { if (t < s) red[t] = fmaxf(red[t], red[t + s]); __syncthreads(); }
    float mx = red[0]; __syncthreads();
    float e0 = __expf(v0 - mx), e1 = __expf(v1 - mx);
    red[t] = e0 + e1; __syncthreads();
    for (int s = 128; s > 0; s >>= 1) { if (t < s) red[t] += red[t + s]; __syncthreads(); }
    float inv = 1.f / red[0];
    sa[t] = e0 * inv; sa[t + 256] = e1 * inv;
    __syncthreads();
  }
  // softmax(z_wm) -> sb
  {
    float v0 = z_wm[t], v1 = z_wm[t + 256];
    red[t] = fmaxf(v0, v1); __syncthreads();
    for (int s = 128; s > 0; s >>= 1) { if (t < s) red[t] = fmaxf(red[t], red[t + s]); __syncthreads(); }
    float mx = red[0]; __syncthreads();
    float e0 = __expf(v0 - mx), e1 = __expf(v1 - mx);
    red[t] = e0 + e1; __syncthreads();
    for (int s = 128; s > 0; s >>= 1) { if (t < s) red[t] += red[t + s]; __syncthreads(); }
    float inv = 1.f / red[0];
    sb[t] = e0 * inv; sb[t + 256] = e1 * inv;
    __syncthreads();
  }
  // softmax(z_wp) -> gates (3 values, every thread computes them)
  float g0 = z_wp[0], g1 = z_wp[1], g2 = z_wp[2];
  float gm = fmaxf(g0, fmaxf(g1, g2));
  float q0 = __expf(g0 - gm), q1 = __expf(g1 - gm), q2 = __expf(g2 - gm);
  float qi = 1.f / (q0 + q1 + q2);
  float a0 = q0 * qi, a1 = q1 * qi, a2 = q2 * qi;

  long base = (long)(blockIdx.x - 48) * 16384 + t;
#pragma unroll 4
  for (int i = 0; i < 64; ++i) {
    long e = base + (long)i * 256;
    int m = (int)(e >> 11);
    int n = (int)(e & (DIM_H - 1));
    memv[e] = a0 * memv[e] + a1 * sa[m] * ca[n] + a2 * sb[m] * cm[n];
  }
}

// ---------------------------------------------------------------------------
// Host-side launch
// ---------------------------------------------------------------------------
extern "C" void kernel_launch(void* const* d_in, const int* in_sizes, int n_in,
                              void* d_out, int out_size, void* d_ws, size_t ws_size,
                              hipStream_t stream) {
  (void)in_sizes; (void)n_in; (void)out_size; (void)ws_size;
  const float* Xa   = (const float*)d_in[0];
  const float* Xm   = (const float*)d_in[1];
  const float* W_ca = (const float*)d_in[2];  const float* b_ca = (const float*)d_in[3];
  const float* W_cm = (const float*)d_in[4];  const float* b_cm = (const float*)d_in[5];
  const float* W_wp = (const float*)d_in[6];  const float* b_wp = (const float*)d_in[7];
  const float* W_wa = (const float*)d_in[8];  const float* b_wa = (const float*)d_in[9];
  const float* W_wm = (const float*)d_in[10]; const float* b_wm = (const float*)d_in[11];
  const float* W_rp = (const float*)d_in[12]; const float* b_rp = (const float*)d_in[13];
  const float* W_r0 = (const float*)d_in[14]; const float* b_r0 = (const float*)d_in[15];
  const float* W_ra = (const float*)d_in[16]; const float* b_ra = (const float*)d_in[17];
  const float* W_rm = (const float*)d_in[18]; const float* b_rm = (const float*)d_in[19];
  float* out = (float*)d_out;

  const int I = DIM_I, H = DIM_H, M = DIM_M, T = DIM_T;

  // ---- workspace carve-up ----
  char* ws = (char*)d_ws;
  size_t off = 0;
  auto take = [&](size_t bytes) -> char* {
    char* p = ws + off;
    off = (off + bytes + 255) & ~(size_t)255;
    return p;
  };
  __bf16* pW_rp = (__bf16*)take((size_t)3 * H * M * 2);
  __bf16* pW_ca = (__bf16*)take((size_t)(H + I) * H * 2);
  __bf16* pW_cm = (__bf16*)take((size_t)(H + I) * H * 2);
  __bf16* pW_wa = (__bf16*)take((size_t)(H + I) * M * 2);
  __bf16* pW_wm = (__bf16*)take((size_t)(H + I) * M * 2);
  __bf16* pW_r0 = (__bf16*)take((size_t)2 * H * H * 2);
  __bf16* pW_ra = (__bf16*)take((size_t)(2 * H + I) * H * 2);
  __bf16* pW_rm = (__bf16*)take((size_t)(2 * H + I) * H * 2);
  __bf16* Xa_bf = (__bf16*)take((size_t)T * I * 2);
  __bf16* Xm_bf = (__bf16*)take((size_t)T * I * 2);
  float*  memv  = (float*) take((size_t)M * H * 4);
  __bf16* stateB = (__bf16*)take((size_t)7 * H * 2);
  __bf16* h_bf[2]  = { stateB,          stateB + H };
  __bf16* ha_bf[2] = { stateB + 2 * H,  stateB + 3 * H };
  __bf16* hm_bf[2] = { stateB + 4 * H,  stateB + 5 * H };
  __bf16* r_bf     =   stateB + 6 * H;
  float* z_rp = (float*)take(512 * 4);
  float* z_wa = (float*)take(512 * 4);
  float* z_wm = (float*)take(512 * 4);
  float* z_wp = (float*)take(4 * 4);
  float* ca_f = (float*)take(2048 * 4);
  float* cm_f = (float*)take(2048 * 4);

  // ---- one-time prep: bf16-pack weights (L2-resident ~115MB), convert X ----
  auto pack = [&](const float* src, __bf16* dst, int K, int N) {
    long total = (long)K * N;
    int grid = (int)((total + 255) / 256);
    pack_weight_kernel<<<grid, 256, 0, stream>>>(src, dst, K, N);
  };
  pack(W_rp, pW_rp, 3 * H, M);
  pack(W_ca, pW_ca, H + I, H);
  pack(W_cm, pW_cm, H + I, H);
  pack(W_wa, pW_wa, H + I, M);
  pack(W_wm, pW_wm, H + I, M);
  pack(W_r0, pW_r0, 2 * H, H);
  pack(W_ra, pW_ra, 2 * H + I, H);
  pack(W_rm, pW_rm, 2 * H + I, H);
  f32_to_bf16_kernel<<<2048, 256, 0, stream>>>(Xa, Xa_bf, (long)T * I);
  f32_to_bf16_kernel<<<2048, 256, 0, stream>>>(Xm, Xm_bf, (long)T * I);
  zero_f32_kernel<<<1024, 256, 0, stream>>>(memv, (long)M * H);
  zero_bf16_kernel<<<56, 256, 0, stream>>>(stateB, (long)7 * H);

  // ---- sequential scan over T steps ----
  for (int t = 0; t < T; ++t) {
    int cur = t & 1, nxt = cur ^ 1;
    const __bf16* xa_t = Xa_bf + (size_t)t * I;
    const __bf16* xm_t = Xm_bf + (size_t)t * I;

    k1_kernel<<<45, 256, 0, stream>>>(
        ha_bf[cur], hm_bf[cur], h_bf[cur], xa_t, xm_t,
        pW_rp, pW_ca, pW_cm, pW_wa, pW_wm,
        b_rp, b_ca, b_cm, b_wa, b_wm,
        W_wp, b_wp,
        z_rp, ca_f, cm_f, z_wa, z_wm, z_wp);

    k2_kernel<<<8, 256, 0, stream>>>(z_rp, memv, r_bf);

    k3_kernel<<<112, 256, 0, stream>>>(
        r_bf, h_bf[cur], ha_bf[cur], hm_bf[cur], xa_t, xm_t,
        pW_r0, pW_ra, pW_rm, b_r0, b_ra, b_rm,
        z_wp, z_wa, z_wm, ca_f, cm_f,
        memv,
        out + (size_t)t * H,
        h_bf[nxt], ha_bf[nxt], hm_bf[nxt]);
  }
}
